// InfiniteNeuralNetwork_50964081934701
// MI455X (gfx1250) — matrix operand order repaired and checked by reading, not compiled, not run on hardware
//
#include <hip/hip_runtime.h>
#include <math.h>

typedef __attribute__((ext_vector_type(16))) _Float16 v16h;
typedef __attribute__((ext_vector_type(8)))  float    v8f;

#define DIM   2048
#define MBLK  128
#define NBLK  128
#define KBLK  32
#define DEPTH 9

// ---------------------------------------------------------------------------
// p[j] = (1/sqrt(D) * prod_{d<9,k<3} cos(inf_w[d,j,k]))^2
// (the reference's circulant scan is exactly this: the state starts uniform,
//  so roll(s,1)-roll(s,-1)==0 bit-exactly and each step is *= cos(angle))
// ---------------------------------------------------------------------------
__global__ void p_kernel(const float* __restrict__ inf_w, float* __restrict__ p) {
    int j = blockIdx.x * blockDim.x + threadIdx.x;
    if (j >= DIM) return;
    float prod = 1.0f;
#pragma unroll
    for (int d = 0; d < DEPTH; ++d) {
        const float* row = inf_w + (size_t)d * DIM * DIM + (size_t)j * DIM;
#pragma unroll
        for (int k = 0; k < 3; ++k) prod *= cosf(row[k]);
    }
    float s0 = prod * (1.0f / sqrtf((float)DIM));
    p[j] = s0 * s0;
}

// ---------------------------------------------------------------------------
// Tiled GEMM: Out = epilogue(A[2048x2048] @ B[2048x2048] + bias (+ p))
// f32 in -> f16 WMMA (f32 accumulate) -> f32 out.
// MODE 0: relu(acc + bias)        MODE 1: tanh(acc + bias + p[col])
//
// LDS tiles are stored in WMMA *fragment order*:
//   A 16x32 f16 fragment, lane L, half h (VGPR v=h/2):
//     M = L%16,  K = 8*(L>=16) + 16*(v>=4) + 2*(v&3) + (h&1)
//   B 32x16 f16 fragment, lane L, half h:
//     N = L%16,  K = 8*(L>=16) + 16*(h>=8) + (h&7)
// so per-lane fragment loads are contiguous 32B reads.
// ---------------------------------------------------------------------------
template <int MODE>
__global__ __launch_bounds__(256)
void gemm_kernel(const float* __restrict__ A, const float* __restrict__ B,
                 const float* __restrict__ bias, const float* __restrict__ p,
                 float* __restrict__ Out) {
    __shared__ __align__(32) _Float16 sA[MBLK * KBLK];   // 8 KB, 8 subtiles of 16x32
    __shared__ __align__(32) _Float16 sB[KBLK * NBLK];   // 8 KB, 8 subtiles of 32x16

    const int tid  = threadIdx.x;
    const int lane = tid & 31;
    const int wave = tid >> 5;          // 0..7
    const int wM   = wave >> 1;         // 0..3  -> 32 rows each
    const int wN   = wave & 1;          // 0..1  -> 64 cols each

    const int blockM = blockIdx.y * MBLK;
    const int blockN = blockIdx.x * NBLK;

    v8f acc[2][4];
#pragma unroll
    for (int tm = 0; tm < 2; ++tm)
#pragma unroll
        for (int tn = 0; tn < 4; ++tn) acc[tm][tn] = (v8f){};

    for (int k0 = 0; k0 < DIM; k0 += KBLK) {
        // ---- stage A tile (128x32) : coalesced float4 loads, swizzled f16 LDS stores
        const float* Ab = A + (size_t)blockM * DIM + k0;
#pragma unroll
        for (int i = 0; i < 4; ++i) {
            int flat = (i * 256 + tid) * 4;              // f32 index in tile
            int m = flat >> 5;
            int k = flat & 31;
            const float4 a4 = *(const float4*)(Ab + (size_t)m * DIM + k);
            const float av[4] = {a4.x, a4.y, a4.z, a4.w};
            int t = m >> 4;
            int Lm = m & 15;
#pragma unroll
            for (int e = 0; e < 4; ++e) {
                int kk = k + e;
                int L = Lm + (((kk >> 3) & 1) << 4);
                int v = (((kk >> 4) & 1) << 2) + ((kk & 7) >> 1);
                int h = (v << 1) + (kk & 1);
                sA[t * 512 + L * 16 + h] = (_Float16)av[e];
            }
        }
        // ---- stage B tile (32x128)
        const float* Bb = B + (size_t)k0 * DIM + blockN;
#pragma unroll
        for (int i = 0; i < 4; ++i) {
            int flat = (i * 256 + tid) * 4;
            int k = flat >> 7;
            int n = flat & 127;
            const float4 b4 = *(const float4*)(Bb + (size_t)k * DIM + n);
            const float bv[4] = {b4.x, b4.y, b4.z, b4.w};
            int hk = ((k >> 4) << 3) + (k & 7);
            int Lhi = ((k >> 3) & 1) ? 16 : 0;
#pragma unroll
            for (int e = 0; e < 4; ++e) {
                int nn = n + e;
                sB[(nn >> 4) * 512 + ((nn & 15) + Lhi) * 16 + hk] = (_Float16)bv[e];
            }
        }
        __syncthreads();

        // ---- fragments + 8 WMMAs
        v16h aF[2], bF[4];
#pragma unroll
        for (int tm = 0; tm < 2; ++tm)
            aF[tm] = *(const v16h*)&sA[(wM * 2 + tm) * 512 + lane * 16];
#pragma unroll
        for (int tn = 0; tn < 4; ++tn)
            bF[tn] = *(const v16h*)&sB[(wN * 4 + tn) * 512 + lane * 16];

#pragma unroll
        for (int tm = 0; tm < 2; ++tm)
#pragma unroll
            for (int tn = 0; tn < 4; ++tn)
                acc[tm][tn] = __builtin_amdgcn_wmma_f32_16x16x32_f16(
                    false, aF[tm], false, bF[tn],
                    (short)0, acc[tm][tn], false, false);

        __syncthreads();
    }

    // ---- epilogue: C/D layout -> row = tm*16 + r + 8*(lane>=16), col = lane%16
    const int rowOff  = (lane >= 16) ? 8 : 0;
    const int col0    = lane & 15;
    const int baseRow = blockM + wM * 32;
    const int baseCol = blockN + wN * 64;
#pragma unroll
    for (int tm = 0; tm < 2; ++tm) {
#pragma unroll
        for (int tn = 0; tn < 4; ++tn) {
            int col = baseCol + tn * 16 + col0;
            float addend = bias[col];
            if (MODE == 1) addend += p[col];
#pragma unroll
            for (int r = 0; r < 8; ++r) {
                int row = baseRow + tm * 16 + r + rowOff;
                float v = acc[tm][tn][r] + addend;
                v = (MODE == 1) ? tanhf(v) : fmaxf(v, 0.0f);
                Out[(size_t)row * DIM + col] = v;
            }
        }
    }
}

// ---------------------------------------------------------------------------
// Launch: 6 GEMMs + 2 tiny p-kernels, stream-ordered.
// inputs: 0:x 1:w0 2:b0 3:w1 4:b1 5:inf_w1 6:cls_w1 7:cls_b1
//         8:w2 9:b2 10:inf_w2 11:cls_w2 12:cls_b2 13:w3 14:b3
// ---------------------------------------------------------------------------
extern "C" void kernel_launch(void* const* d_in, const int* in_sizes, int n_in,
                              void* d_out, int out_size, void* d_ws, size_t ws_size,
                              hipStream_t stream) {
    const float* x      = (const float*)d_in[0];
    const float* w0     = (const float*)d_in[1];
    const float* b0     = (const float*)d_in[2];
    const float* w1     = (const float*)d_in[3];
    const float* b1     = (const float*)d_in[4];
    const float* inf_w1 = (const float*)d_in[5];
    const float* cls_w1 = (const float*)d_in[6];
    const float* cls_b1 = (const float*)d_in[7];
    const float* w2     = (const float*)d_in[8];
    const float* b2     = (const float*)d_in[9];
    const float* inf_w2 = (const float*)d_in[10];
    const float* cls_w2 = (const float*)d_in[11];
    const float* cls_b2 = (const float*)d_in[12];
    const float* w3     = (const float*)d_in[13];
    const float* b3     = (const float*)d_in[14];
    float* out = (float*)d_out;

    float* act0 = (float*)d_ws;                 // 2048*2048 f32
    float* act1 = act0 + (size_t)DIM * DIM;     // 2048*2048 f32
    float* p1   = act1 + (size_t)DIM * DIM;     // 2048 f32
    float* p2   = p1 + DIM;                     // 2048 f32

    p_kernel<<<DIM / 256, 256, 0, stream>>>(inf_w1, p1);
    p_kernel<<<DIM / 256, 256, 0, stream>>>(inf_w2, p2);

    dim3 grid(DIM / NBLK, DIM / MBLK);          // 16 x 16
    dim3 block(256);

    // h0 = relu(x @ w0 + b0)
    gemm_kernel<0><<<grid, block, 0, stream>>>(x, w0, b0, nullptr, act0);
    // h1 = relu(h0 @ w1 + b1)
    gemm_kernel<0><<<grid, block, 0, stream>>>(act0, w1, b1, nullptr, act1);
    // h2 = tanh(h1 @ cls_w1 + cls_b1 + p1)
    gemm_kernel<1><<<grid, block, 0, stream>>>(act1, cls_w1, cls_b1, p1, act0);
    // h3 = relu(h2 @ w2 + b2)
    gemm_kernel<0><<<grid, block, 0, stream>>>(act0, w2, b2, nullptr, act1);
    // h4 = tanh(h3 @ cls_w2 + cls_b2 + p2)
    gemm_kernel<1><<<grid, block, 0, stream>>>(act1, cls_w2, cls_b2, p2, act0);
    // out = relu(h4 @ w3 + b3)
    gemm_kernel<0><<<grid, block, 0, stream>>>(act0, w3, b3, nullptr, out);
}
